// CalculateVector_45930380264074
// MI455X (gfx1250) — matrix operand order repaired and checked by reading, not compiled, not run on hardware
//
#include <hip/hip_runtime.h>

typedef _Float16     v16h __attribute__((ext_vector_type(16)));
typedef _Float16     v8h  __attribute__((ext_vector_type(8)));
typedef float        v8f  __attribute__((ext_vector_type(8)));
typedef unsigned int v4u  __attribute__((ext_vector_type(4)));

#define NPIX (270 * 480)      /* 129600 pixels */
#define K2 16
#define ROW 800               /* 50*16 fp16 per pixel per array */
#define WAVES_PER_BLOCK 8
#define PPW 8                 /* pixels per wave */

/* Center-out spiral order of the 7x7 grid (reference _spiral_order(3)):
   SPIRAL49[rank] = flat grid index. */
__device__ __constant__ int SPIRAL49[49] = {
  24,25,32,31,30,23,16,
  17,18,19,26,33,40,39,
  38,37,36,29,22,15, 8,
   9,10,11,12,13,20,27,
  34,41,48,47,46,45,44,
  43,42,35,28,21,14, 7,
   0, 1, 2, 3, 4, 5, 6 };

/* |a-b| for exact-integer fp16: packed subtract, clear sign bits. */
__device__ __forceinline__ v8h absdiff8(v8h a, v8h b) {
  v8h d = a - b;
  v4u u;
  __builtin_memcpy(&u, &d, 16);
  u = u & 0x7FFF7FFFu;
  __builtin_memcpy(&d, &u, 16);
  return d;
}

__global__ __launch_bounds__(256) void sad_spiral_argmin_kernel(
    const _Float16* __restrict__ w1, const _Float16* __restrict__ w2,
    float* __restrict__ out)
{
  const int ln   = threadIdx.x & 31;
  const int wid  = blockIdx.x * WAVES_PER_BLOCK + (threadIdx.x >> 5);
  const int m    = ln & 15;      // C column N
  const int hi   = ln >> 4;      // which half of the 16 SAD elements
  const int e0   = hi * 8;

  /* ---- per-wave setup, amortized over PPW pixels ---- */

  // Spiral permutation registers for the post-reduction rank->idx lookup.
  const int sp_lo = SPIRAL49[ln];
  const int sp_hi = SPIRAL49[(ln + 32 < 49) ? (ln + 32) : 48];

  // Spiral-permuted memory candidate for each slot this lane supplies.
  // slot s (s<49) -> memory row SPIRAL[s]; slot 49 -> input-MV row 49;
  // slots 50..63 are dead -> duplicate row 49 (stays in L0).
  const int memA0 = SPIRAL49[m];                         // slots  0..15
  const int memB0 = SPIRAL49[m + 16];                    // slots 16..31
  const int memA1 = SPIRAL49[m + 32];                    // slots 32..47
  const int memB1 = (m == 0) ? SPIRAL49[48] : 49;        // slots 48..63

  // Element offsets (lane-invariant across pixels).
  const int offA0 = memA0 * K2 + e0;
  const int offB0 = memB0 * K2 + e0;
  const int offA1 = memA1 * K2 + e0;
  const int offB1 = memB1 * K2 + e0;

  // B selector: B[k,0]=1 for k<16, B[k,1]=1 for k>=16, else 0.
  // (16-bit B 32x16: lane holds column N=lane%16; elements s<8 are k<16, s>=8 are k>=16.)
  v16h bf;
#pragma unroll
  for (int i = 0; i < 16; ++i)
    bf[i] = ((m == 0 && i < 8) || (m == 1 && i >= 8)) ? (_Float16)1 : (_Float16)0;

  const int  N      = m;
  const int  Mb     = hi * 8;
  const bool laneOK = (N <= 1);

  float* outVec  = out;
  float* outTmpl = out + 2  * (size_t)NPIX;
  float* outMask = out + 18 * (size_t)NPIX;
  float* outCost = out + 19 * (size_t)NPIX;

  /* ---- pixel loop ---- */
  const int pixBase = wid * PPW;
  for (int p = 0; p < PPW; ++p) {
    const int pix = pixBase + p;
    const _Float16* w1p = w1 + (size_t)pix * ROW;
    const _Float16* w2p = w2 + (size_t)pix * ROW;

    // Stream ahead ~2 pixels (3.2KB/array) -> global_prefetch_b8
    __builtin_prefetch((const char*)w1p + 2 * ROW * 2 + ln * 64, 0, 1);
    __builtin_prefetch((const char*)w2p + 2 * ROW * 2 + ln * 64, 0, 1);

    // ---- group 0: slots 0..31 ----
    v8h dA = absdiff8(*(const v8h*)(w1p + offA0), *(const v8h*)(w2p + offA0));
    v8h dB = absdiff8(*(const v8h*)(w1p + offB0), *(const v8h*)(w2p + offB0));
    v16h af0;
#pragma unroll
    for (int i = 0; i < 8; ++i) { af0[i] = dA[i]; af0[8 + i] = dB[i]; }

    // ---- group 1: slots 32..63 ----
    dA = absdiff8(*(const v8h*)(w1p + offA1), *(const v8h*)(w2p + offA1));
    dB = absdiff8(*(const v8h*)(w1p + offB1), *(const v8h*)(w2p + offB1));
    v16h af1;
#pragma unroll
    for (int i = 0; i < 8; ++i) { af1[i] = dA[i]; af1[8 + i] = dB[i]; }

    v8f c0 = {};
    v8f c1 = {};
    c0 = __builtin_amdgcn_wmma_f32_16x16x32_f16(false, af0, false, bf,
                                                (short)0, c0, false, false);
    c1 = __builtin_amdgcn_wmma_f32_16x16x32_f16(false, af1, false, bf,
                                                (short)0, c1, false, false);

    // ---- branch-free key build in float: key = cost*64 + slot (exact) ----
    float kf = 1.0e9f;
#pragma unroll
    for (int r = 0; r < 8; ++r) {
      const int s0 = 16 * N + Mb + r;            // 0..31 when laneOK
      const float v0 = fmaf(c0[r], 64.0f, (float)s0);
      kf = fminf(kf, laneOK ? v0 : 1.0e9f);
      const int s1 = 32 + 16 * N + Mb + r;       // 32..63
      const bool ok1 = laneOK && (s1 < 49);      // slot 49 = input MV, excluded
      const float v1 = fmaf(c1[r], 64.0f, (float)s1);
      kf = fminf(kf, ok1 ? v1 : 1.0e9f);
    }

    // Input-MV cost: slot 49 -> C[M=1,N=1] -> lane 1, c1[1].
    const int cost49 = __shfl((int)c1[1], 1, 32);

    // wave32 float-min reduction
#pragma unroll
    for (int off = 16; off >= 1; off >>= 1)
      kf = fminf(kf, __shfl_xor(kf, off, 32));

    const int ki      = (int)kf;
    const int rank    = ki & 63;                 // spiral rank of winner
    const int cost_bm = ki >> 6;

    // rank -> flat grid index via register-resident SPIRAL (2 shuffles)
    const int r5   = rank & 31;
    const int vlo  = __shfl(sp_lo, r5, 32);
    const int vhi  = __shfl(sp_hi, r5, 32);
    const int idx_bm = (rank < 32) ? vlo : vhi;

    const bool mvmask  = cost49 < cost_bm;
    const int  min_idx  = mvmask ? 49 : idx_bm;
    const int  min_cost = mvmask ? cost49 : cost_bm;

    // ---- outputs: vector[P][2] | templates[P][16] | mask[P] | cost[P] ----
    if (ln < 16)
      outTmpl[(size_t)pix * 16 + ln] = (float)w1p[min_idx * K2 + ln];
    if (ln == 0) {
      outVec[(size_t)pix * 2 + 0] = (float)(3 - idx_bm / 7);
      outVec[(size_t)pix * 2 + 1] = (float)(3 - idx_bm % 7);
      outMask[pix] = mvmask ? 1.0f : 0.0f;
      outCost[pix] = (float)min_cost;
    }
  }
}

extern "C" void kernel_launch(void* const* d_in, const int* in_sizes, int n_in,
                              void* d_out, int out_size, void* d_ws, size_t ws_size,
                              hipStream_t stream) {
  (void)in_sizes; (void)n_in; (void)out_size; (void)d_ws; (void)ws_size;
  const _Float16* w1 = (const _Float16*)d_in[0];
  const _Float16* w2 = (const _Float16*)d_in[1];
  float* out = (float*)d_out;
  dim3 block(WAVES_PER_BLOCK * 32);                    // 256 threads = 8 wave32
  dim3 grid(NPIX / (WAVES_PER_BLOCK * PPW));           // 2025 blocks, exact cover
  sad_spiral_argmin_kernel<<<grid, block, 0, stream>>>(w1, w2, out);
}
